// EdgePrompt_plus_83176336654763
// MI455X (gfx1250) — matrix operand order repaired and checked by reading, not compiled, not run on hardware
//
#include <hip/hip_runtime.h>
#include <hip/hip_fp16.h>

typedef __attribute__((ext_vector_type(16))) _Float16 v16h;
typedef __attribute__((ext_vector_type(8)))  float    v8f;

#define IN_DIM 128
#define TWO_D  256
#define N_ANCH 5

__device__ __forceinline__ v16h zero16() {
    v16h v;
#pragma unroll
    for (int i = 0; i < 16; ++i) v[i] = (_Float16)0.0f;
    return v;
}

__device__ __forceinline__ v16h pack16(float4 a, float4 b, float4 c, float4 d) {
    v16h v;
    v[0]=(_Float16)a.x; v[1]=(_Float16)a.y; v[2]=(_Float16)a.z; v[3]=(_Float16)a.w;
    v[4]=(_Float16)b.x; v[5]=(_Float16)b.y; v[6]=(_Float16)b.z; v[7]=(_Float16)b.w;
    v[8]=(_Float16)c.x; v[9]=(_Float16)c.y; v[10]=(_Float16)c.z; v[11]=(_Float16)c.w;
    v[12]=(_Float16)d.x; v[13]=(_Float16)d.y; v[14]=(_Float16)d.z; v[15]=(_Float16)d.w;
    return v;
}

__global__ void zero_ws_kernel(float* __restrict__ p, long long n) {
    long long i = (long long)blockIdx.x * blockDim.x + threadIdx.x;
    if (i < n) p[i] = 0.0f;
}

// One wave32 handles 16 edges.
//   logitsT = Wpad(16x256) x featT(256x16)  -> lane n holds edge n's logits
//   edge_prompt = probs(16x32) x APt(32x16) per 16-col tile -> atomic scatter
__global__ __launch_bounds__(256)
void edge_wmma_kernel(const float* __restrict__ x,
                      const int*   __restrict__ ei,
                      const float* __restrict__ W,
                      const float* __restrict__ b_lin,
                      const float* __restrict__ ap,
                      float* __restrict__ acc,
                      float* __restrict__ deg,
                      int E, int ntiles)
{
    const int lane = threadIdx.x & 31;
    const int wave = threadIdx.x >> 5;
    const int tile = blockIdx.x * (blockDim.x >> 5) + wave;
    if (tile >= ntiles) return;          // uniform across the wave

    const int half = lane >> 4;          // 0: lanes 0-15, 1: lanes 16-31
    const int lr   = lane & 15;

    // ---- Preload W into f16 A-matrix layout: 8 K-steps of 32 ----
    // lane<16 : VGPR0-3 = K k0+0..7, VGPR4-7 = K k0+16..23
    // lane>=16: VGPR0-3 = K k0+8..15, VGPR4-7 = K k0+24..31
    v16h Wt[8];
    const int offA = half * 8;
#pragma unroll
    for (int ks = 0; ks < 8; ++ks) {
        const int k0 = ks * 32;
        v16h w = zero16();
        if (lr < N_ANCH) {
            const float* wp = W + lr * TWO_D + k0 + offA;
            float4 a0 = *(const float4*)(wp);
            float4 a1 = *(const float4*)(wp + 4);
            float4 a2 = *(const float4*)(wp + 16);
            float4 a3 = *(const float4*)(wp + 20);
            w = pack16(a0, a1, a2, a3);
        }
        Wt[ks] = w;
    }

    // ---- Preload anchor_prompt as B-matrix tiles (32x16), 8 col-tiles ----
    // lane<16 holds K=0..15 of column (t*16+lr); only K<5 nonzero
    v16h APt[8];
#pragma unroll
    for (int t = 0; t < 8; ++t) {
        v16h bv = zero16();
        if (half == 0) {
            const int col = t * 16 + lr;
            bv[0] = (_Float16)ap[0 * IN_DIM + col];
            bv[1] = (_Float16)ap[1 * IN_DIM + col];
            bv[2] = (_Float16)ap[2 * IN_DIM + col];
            bv[3] = (_Float16)ap[3 * IN_DIM + col];
            bv[4] = (_Float16)ap[4 * IN_DIM + col];
        }
        APt[t] = bv;
    }

    // ---- Edge indices: lane n (and n+16) owns edge tile*16 + lr ----
    const int e    = tile * 16 + lr;
    const int srci = ei[e];
    const int dsti = ei[E + e];
    const float* srow = x + (long long)srci * IN_DIM;
    const float* drow = x + (long long)dsti * IN_DIM;

    // ---- GEMM 1: logitsT accumulate over 8 K-steps ----
    v8f c = {};
    const int off2 = half * 8;           // B layout: lane<16 K 0..7,16..23, lane>=16 K 8..15,24..31
#pragma unroll
    for (int ks = 0; ks < 8; ++ks) {
        const int kk = ks * 32 + off2;   // feature idx kk..kk+7, kk+16..kk+23, never straddles 128
        const float* fp = (kk < IN_DIM) ? (srow + kk) : (drow + (kk - IN_DIM));
        float4 f0 = *(const float4*)(fp);
        float4 f1 = *(const float4*)(fp + 4);
        float4 f2 = *(const float4*)(fp + 16);
        float4 f3 = *(const float4*)(fp + 20);
        v16h bf = pack16(f0, f1, f2, f3);
        c = __builtin_amdgcn_wmma_f32_16x16x32_f16(false, Wt[ks], false, bf,
                                                   (short)0, c, false, false);
    }

    // ---- Per-lane: bias + leaky_relu + softmax over 5 anchors ----
    float p[N_ANCH];
    float mx = -3.0e38f;
#pragma unroll
    for (int a = 0; a < N_ANCH; ++a) {
        float l = c[a] + b_lin[a];
        l = (l > 0.0f) ? l : 0.01f * l;
        p[a] = l;
        mx = fmaxf(mx, l);
    }
    float s = 0.0f;
#pragma unroll
    for (int a = 0; a < N_ANCH; ++a) { p[a] = __expf(p[a] - mx); s += p[a]; }
    const float inv = 1.0f / s;
#pragma unroll
    for (int a = 0; a < N_ANCH; ++a) p[a] *= inv;

    // ---- A2: probs as 16x32 f16 A-matrix (K = anchors, padded) ----
    v16h pa = zero16();
    if (half == 0) {
        pa[0] = (_Float16)p[0];
        pa[1] = (_Float16)p[1];
        pa[2] = (_Float16)p[2];
        pa[3] = (_Float16)p[3];
        pa[4] = (_Float16)p[4];
    }

    // ---- Broadcast dst indices: D-row r of half h belongs to edge 8h+r ----
    int dsts[8];
#pragma unroll
    for (int r = 0; r < 8; ++r) dsts[r] = __shfl(dsti, half * 8 + r, 32);

    // ---- GEMM 2 + scatter-add per 16-column tile ----
#pragma unroll
    for (int t = 0; t < 8; ++t) {
        v8f d = {};
        d = __builtin_amdgcn_wmma_f32_16x16x32_f16(false, pa, false, APt[t],
                                                   (short)0, d, false, false);
        const int col = t * 16 + lr;
#pragma unroll
        for (int r = 0; r < 8; ++r) {
            atomicAdd(acc + (long long)dsts[r] * IN_DIM + col, d[r]);
        }
    }

    // ---- Degree counts (one per edge) ----
    if (lane < 16) atomicAdd(deg + dsti, 1.0f);
}

// Scalar fallback for E % 16 edges (not launched when E divisible by 16).
__global__ void edge_tail_kernel(const float* __restrict__ x,
                                 const int*   __restrict__ ei,
                                 const float* __restrict__ W,
                                 const float* __restrict__ b_lin,
                                 const float* __restrict__ ap,
                                 float* __restrict__ acc,
                                 float* __restrict__ deg,
                                 int E, int start, int rem)
{
    int i = blockIdx.x * blockDim.x + threadIdx.x;
    if (i >= rem) return;
    int e = start + i;
    int s = ei[e], dd = ei[E + e];
    const float* srow = x + (long long)s * IN_DIM;
    const float* drow = x + (long long)dd * IN_DIM;
    float lg[N_ANCH];
    float mx = -3.0e38f;
    for (int a = 0; a < N_ANCH; ++a) {
        float acc0 = 0.0f;
        const float* wr = W + a * TWO_D;
        for (int k = 0; k < IN_DIM; ++k) acc0 += srow[k] * wr[k];
        for (int k = 0; k < IN_DIM; ++k) acc0 += drow[k] * wr[IN_DIM + k];
        float l = acc0 + b_lin[a];
        l = (l > 0.0f) ? l : 0.01f * l;
        lg[a] = l; mx = fmaxf(mx, l);
    }
    float ssum = 0.0f;
    for (int a = 0; a < N_ANCH; ++a) { lg[a] = __expf(lg[a] - mx); ssum += lg[a]; }
    for (int a = 0; a < N_ANCH; ++a) lg[a] /= ssum;
    for (int cidx = 0; cidx < IN_DIM; ++cidx) {
        float v = 0.0f;
        for (int a = 0; a < N_ANCH; ++a) v += lg[a] * ap[a * IN_DIM + cidx];
        atomicAdd(acc + (long long)dd * IN_DIM + cidx, v);
    }
    atomicAdd(deg + dd, 1.0f);
}

__global__ void finalize_kernel(const float* __restrict__ x,
                                const float* __restrict__ acc,
                                const float* __restrict__ deg,
                                float* __restrict__ out,
                                long long total)
{
    long long i = (long long)blockIdx.x * blockDim.x + threadIdx.x;
    if (i < total) {
        int node = (int)(i >> 7);     // / IN_DIM
        float dg = fmaxf(deg[node], 1.0f);
        out[i] = x[i] + acc[i] / dg;
    }
}

extern "C" void kernel_launch(void* const* d_in, const int* in_sizes, int n_in,
                              void* d_out, int out_size, void* d_ws, size_t ws_size,
                              hipStream_t stream) {
    const float* x     = (const float*)d_in[0];
    const int*   ei    = (const int*)d_in[1];
    const float* W     = (const float*)d_in[2];
    const float* b_lin = (const float*)d_in[3];
    const float* ap    = (const float*)d_in[4];
    float* out = (float*)d_out;

    const int n_nodes = in_sizes[0] / IN_DIM;
    const int E       = in_sizes[1] / 2;

    float* acc = (float*)d_ws;                              // n_nodes * 128
    float* deg = acc + (size_t)n_nodes * IN_DIM;            // n_nodes

    // Zero accumulator + degree
    const long long nz = (long long)n_nodes * IN_DIM + n_nodes;
    zero_ws_kernel<<<(int)((nz + 255) / 256), 256, 0, stream>>>(acc, nz);

    // Main WMMA edge kernel: one wave32 per 16-edge tile, 8 waves per block
    const int ntiles = E / 16;
    if (ntiles > 0) {
        const int wpb = 8;
        const int nblk = (ntiles + wpb - 1) / wpb;
        edge_wmma_kernel<<<nblk, wpb * 32, 0, stream>>>(x, ei, W, b_lin, ap,
                                                        acc, deg, E, ntiles);
    }
    const int rem = E % 16;
    if (rem > 0) {
        edge_tail_kernel<<<(rem + 63) / 64, 64, 0, stream>>>(x, ei, W, b_lin, ap,
                                                             acc, deg, E, E - rem, rem);
    }

    // out = x + acc / max(deg, 1)
    const long long total = (long long)n_nodes * IN_DIM;
    finalize_kernel<<<(int)((total + 255) / 256), 256, 0, stream>>>(x, acc, deg, out, total);
}
